// LinearAttention_50680614093003
// MI455X (gfx1250) — matrix operand order repaired and checked by reading, not compiled, not run on hardware
//
#include <hip/hip_runtime.h>

// ---- types for WMMA fragments (gfx1250, wave32) ----
typedef __attribute__((ext_vector_type(16))) __bf16        v16bf;
typedef __attribute__((ext_vector_type(8)))  float         v8f;
typedef __attribute__((ext_vector_type(4)))  unsigned int  u32x4;

union Frag {            // 32 bytes: one 16x16x32 bf16 A or B operand per lane
    v16bf v;
    u32x4 q[2];
    unsigned short u[16];
};

#define B_      8
#define N_      8192
#define D_      256
#define HEADS   8
#define DH      64
#define INNER   512
#define NCHUNK  16
#define KC      (N_ / NCHUNK)   // 512
#define OPITCH  520             // LDS row pitch (halves): 1040B, 16B-aligned,
                                // 260 dwords -> rows stagger 4 banks (no 16-way conflicts)

__device__ __forceinline__ unsigned short f2bf(float f) {
    unsigned int u = __float_as_uint(f);
    unsigned int r = (u + 0x7FFFu + ((u >> 16) & 1u)) >> 16;   // RNE
    return (unsigned short)r;
}

__device__ __forceinline__ v8f vzero8() {
    v8f z = {0.f, 0.f, 0.f, 0.f, 0.f, 0.f, 0.f, 0.f};
    return z;
}

// ---------------- kernel 0: fp32 -> bf16 ----------------
__global__ void cvt_bf16(const float* __restrict__ in,
                         unsigned short* __restrict__ out, int n) {
    int i = blockIdx.x * blockDim.x + threadIdx.x;
    int stride = gridDim.x * blockDim.x;
    for (; i < n; i += stride) out[i] = f2bf(in[i]);
}

// ---------------- kernel 1: qkv projection + instance norm ----------------
// grid = (B*N/64, 24); block = 128 (4 waves). Strip s covers qkv cols [64s,64s+64).
__global__ __launch_bounds__(128) void qkv_norm(
    const unsigned short* __restrict__ xb,   // [B*N, 256] bf16
    const unsigned short* __restrict__ wb,   // [1536, 256] bf16
    unsigned short* __restrict__ qb,         // [B*H, N, 64]
    unsigned short* __restrict__ kT,         // [B*H, 64, N]
    unsigned short* __restrict__ vT)         // [B*H, 64, N]
{
    __shared__ float tile[64][65];
    const int tid  = threadIdx.x;
    const int wave = tid >> 5, lane = tid & 31;
    const int lh   = lane >> 4, ln = lane & 15;
    const long row0  = (long)blockIdx.x * 64;     // flat row over B*N
    const int  strip = blockIdx.y;                // 0..23
    const int  col0  = strip * 64;

    const unsigned short* arow = xb + (row0 + wave * 16 + ln) * D_;
    v8f c[4];
    #pragma unroll
    for (int t = 0; t < 4; ++t) c[t] = vzero8();

    #pragma unroll
    for (int kit = 0; kit < D_ / 32; ++kit) {
        const int kb = kit * 32;
        Frag a;
        a.q[0] = *(const u32x4*)(arow + kb + lh * 8);
        a.q[1] = *(const u32x4*)(arow + kb + lh * 8 + 16);
        #pragma unroll
        for (int t = 0; t < 4; ++t) {
            const unsigned short* brow = wb + (long)(col0 + t * 16 + ln) * D_ + kb + lh * 8;
            Frag b;
            b.q[0] = *(const u32x4*)(brow);
            b.q[1] = *(const u32x4*)(brow + 16);
            c[t] = __builtin_amdgcn_wmma_f32_16x16x32_bf16(
                false, a.v, false, b.v, (short)0, c[t], false, false);
        }
    }

    #pragma unroll
    for (int t = 0; t < 4; ++t)
        #pragma unroll
        for (int r = 0; r < 8; ++r)
            tile[wave * 16 + r + lh * 8][t * 16 + ln] = c[t][r];
    __syncthreads();

    const long b  = row0 >> 13;          // / N_
    const long n0 = row0 & (N_ - 1);

    if (strip < 8) {                      // ---- q: straight bf16 store
        const int h = strip;
        unsigned short* dst = qb + ((b * HEADS + h) * (long)N_ + n0) * DH;
        for (int idx = tid; idx < 64 * 64; idx += 128) {
            const int rr = idx >> 6, d = idx & 63;
            dst[(long)rr * DH + d] = f2bf(tile[rr][d]);
        }
    } else {                              // ---- k/v: instance-norm over d, store transposed
        const int isv = (strip >= 16);
        const int h   = strip - (isv ? 16 : 8);
        unsigned short* dst = (isv ? vT : kT) + ((b * HEADS + h) * (long)DH) * N_ + n0;
        if (tid < 64) {
            float mu = 0.f;
            #pragma unroll
            for (int d = 0; d < 64; ++d) mu += tile[tid][d];
            mu *= (1.0f / 64.0f);
            float var = 0.f;
            #pragma unroll
            for (int d = 0; d < 64; ++d) { float t = tile[tid][d] - mu; var += t * t; }
            var *= (1.0f / 64.0f);
            const float rs = rsqrtf(var + 1e-5f);
            #pragma unroll
            for (int d = 0; d < 64; ++d)
                dst[(long)d * N_ + tid] = f2bf((tile[tid][d] - mu) * rs);
        }
    }
}

// ---------------- kernel 2: dots partial = k^T v over a K chunk ----------------
// grid = (B*H, NCHUNK); block = 128 (4 waves, one 16-row d-strip each).
__global__ __launch_bounds__(128) void dots_partial(
    const unsigned short* __restrict__ kT,   // [B*H, 64, N]
    const unsigned short* __restrict__ vT,   // [B*H, 64, N]
    float* __restrict__ part)                // [B*H, NCHUNK, 64, 64]
{
    const int tid  = threadIdx.x;
    const int wave = tid >> 5, lane = tid & 31;
    const int lh   = lane >> 4, ln = lane & 15;
    const int bh    = blockIdx.x;
    const int chunk = blockIdx.y;
    const long nb   = (long)chunk * KC;
    const int d0    = wave * 16;

    const unsigned short* arow = kT + ((long)bh * DH + d0 + ln) * N_ + nb;
    v8f c[4];
    #pragma unroll
    for (int t = 0; t < 4; ++t) c[t] = vzero8();

    for (int kit = 0; kit < KC / 32; ++kit) {
        const int kb = kit * 32;
        Frag a;
        a.q[0] = *(const u32x4*)(arow + kb + lh * 8);
        a.q[1] = *(const u32x4*)(arow + kb + lh * 8 + 16);
        #pragma unroll
        for (int t = 0; t < 4; ++t) {
            const unsigned short* brow =
                vT + ((long)bh * DH + t * 16 + ln) * N_ + nb + kb + lh * 8;
            Frag b;
            b.q[0] = *(const u32x4*)(brow);
            b.q[1] = *(const u32x4*)(brow + 16);
            c[t] = __builtin_amdgcn_wmma_f32_16x16x32_bf16(
                false, a.v, false, b.v, (short)0, c[t], false, false);
        }
    }

    float* dst = part + (((long)bh * NCHUNK + chunk) << 12);
    #pragma unroll
    for (int t = 0; t < 4; ++t)
        #pragma unroll
        for (int r = 0; r < 8; ++r)
            dst[(d0 + r + lh * 8) * 64 + t * 16 + ln] = c[t][r];
}

// ---------------- kernel 2b: deterministic chunk reduce, scale 1/N, store dots^T bf16 ----
__global__ void dots_reduce(const float* __restrict__ part,
                            unsigned short* __restrict__ dotsT) {
    const int idx = blockIdx.x * blockDim.x + threadIdx.x;  // < B*H*4096
    const int bh  = idx >> 12;
    const int rem = idx & 4095;
    const int d = rem >> 6, e = rem & 63;
    float s = 0.f;
    #pragma unroll
    for (int ch = 0; ch < NCHUNK; ++ch)
        s += part[(((long)bh * NCHUNK + ch) << 12) + rem];
    dotsT[((long)bh << 12) + e * 64 + d] = f2bf(s * (1.0f / (float)N_));
}

// ---------------- kernel 3: out = (q @ dots) @ w_out^T + b ----------------
// grid = B*N/64; block = 128 (4 waves). Fused: per-head 64x64 GEMM -> LDS bf16 tile
// -> 64x256x512 GEMM against w_out^T -> bias -> fp32 output.
__global__ __launch_bounds__(128) void out_proj(
    const unsigned short* __restrict__ qb,     // [B*H, N, 64]
    const unsigned short* __restrict__ dotsT,  // [B*H, 64(e), 64(d)]
    const unsigned short* __restrict__ woutb,  // [256, 512]
    const float* __restrict__ bias,            // [256]
    float* __restrict__ out)                   // [B*N, 256]
{
    __shared__ unsigned short O[64][OPITCH];   // ~65 KB bf16 tile, bank-staggered rows
    const int tid  = threadIdx.x;
    const int wave = tid >> 5, lane = tid & 31;
    const int lh   = lane >> 4, ln = lane & 15;
    const long row0 = (long)blockIdx.x * 64;
    const long b    = row0 >> 13;
    const long n0   = row0 & (N_ - 1);

    // ---- stage 1: O[n, h*64+e] = sum_d q[n,d] * dots[d,e]
    for (int h = 0; h < HEADS; ++h) {
        const long bh = b * HEADS + h;
        const unsigned short* arow = qb + (bh * N_ + n0 + wave * 16 + ln) * DH;
        v8f c[4];
        #pragma unroll
        for (int t = 0; t < 4; ++t) c[t] = vzero8();
        #pragma unroll
        for (int kit = 0; kit < 2; ++kit) {
            const int kb = kit * 32;
            Frag a;
            a.q[0] = *(const u32x4*)(arow + kb + lh * 8);
            a.q[1] = *(const u32x4*)(arow + kb + lh * 8 + 16);
            #pragma unroll
            for (int t = 0; t < 4; ++t) {
                const unsigned short* brow = dotsT + (bh << 12) + (t * 16 + ln) * 64 + kb + lh * 8;
                Frag bb;
                bb.q[0] = *(const u32x4*)(brow);
                bb.q[1] = *(const u32x4*)(brow + 16);
                c[t] = __builtin_amdgcn_wmma_f32_16x16x32_bf16(
                    false, a.v, false, bb.v, (short)0, c[t], false, false);
            }
        }
        #pragma unroll
        for (int t = 0; t < 4; ++t)
            #pragma unroll
            for (int r = 0; r < 8; ++r)
                O[wave * 16 + r + lh * 8][h * 64 + t * 16 + ln] = f2bf(c[t][r]);
    }
    __syncthreads();

    // ---- stage 2: out[n, d'] = sum_e O[n,e] * w_out[d',e] + b[d']
    v8f acc[16];
    #pragma unroll
    for (int t = 0; t < 16; ++t) acc[t] = vzero8();

    for (int kit = 0; kit < INNER / 32; ++kit) {
        const int kb = kit * 32;
        Frag a;
        a.q[0] = *(const u32x4*)(&O[wave * 16 + ln][kb + lh * 8]);
        a.q[1] = *(const u32x4*)(&O[wave * 16 + ln][kb + lh * 8 + 16]);
        #pragma unroll
        for (int t = 0; t < 16; ++t) {
            const unsigned short* brow = woutb + (long)(t * 16 + ln) * INNER + kb + lh * 8;
            Frag bb;
            bb.q[0] = *(const u32x4*)(brow);
            bb.q[1] = *(const u32x4*)(brow + 16);
            acc[t] = __builtin_amdgcn_wmma_f32_16x16x32_bf16(
                false, a.v, false, bb.v, (short)0, acc[t], false, false);
        }
    }

    #pragma unroll
    for (int t = 0; t < 16; ++t) {
        const int col = t * 16 + ln;
        const float bv = bias[col];
        #pragma unroll
        for (int r = 0; r < 8; ++r) {
            const long m = wave * 16 + r + lh * 8;
            out[(row0 + m) * D_ + col] = acc[t][r] + bv;
        }
    }
}

// ---------------- host side ----------------
extern "C" void kernel_launch(void* const* d_in, const int* in_sizes, int n_in,
                              void* d_out, int out_size, void* d_ws, size_t ws_size,
                              hipStream_t stream) {
    (void)in_sizes; (void)n_in; (void)out_size; (void)ws_size;
    const float* x     = (const float*)d_in[0];   // [8,8192,256]
    const float* w_qkv = (const float*)d_in[1];   // [1536,256]
    const float* w_out = (const float*)d_in[2];   // [256,512]
    const float* b_out = (const float*)d_in[3];   // [256]
    float* out = (float*)d_out;

    unsigned char* w = (unsigned char*)d_ws;
    size_t o = 0;
    unsigned short* xb    = (unsigned short*)(w + o); o += (size_t)B_ * N_ * D_ * 2;       // 33.5 MB
    unsigned short* wqkvb = (unsigned short*)(w + o); o += (size_t)1536 * 256 * 2;         // 0.75 MB
    unsigned short* woutb = (unsigned short*)(w + o); o += (size_t)256 * 512 * 2;          // 0.25 MB
    unsigned short* qb    = (unsigned short*)(w + o); o += (size_t)B_ * HEADS * N_ * DH * 2; // 64 MB
    unsigned short* kT    = (unsigned short*)(w + o); o += (size_t)B_ * HEADS * N_ * DH * 2;
    unsigned short* vT    = (unsigned short*)(w + o); o += (size_t)B_ * HEADS * N_ * DH * 2;
    float*          part  = (float*)(w + o);          o += (size_t)B_ * HEADS * NCHUNK * 4096 * 4; // 16 MB
    unsigned short* dotsT = (unsigned short*)(w + o); o += (size_t)B_ * HEADS * 4096 * 2;

    cvt_bf16<<<4096, 256, 0, stream>>>(x,     xb,    B_ * N_ * D_);
    cvt_bf16<<<192,  256, 0, stream>>>(w_qkv, wqkvb, 1536 * 256);
    cvt_bf16<<<96,   256, 0, stream>>>(w_out, woutb, 256 * 512);

    qkv_norm<<<dim3((B_ * N_) / 64, 24), 128, 0, stream>>>(xb, wqkvb, qb, kT, vT);
    dots_partial<<<dim3(B_ * HEADS, NCHUNK), 128, 0, stream>>>(kT, vT, part);
    dots_reduce<<<(B_ * HEADS * 4096) / 256, 256, 0, stream>>>(part, dotsT);
    out_proj<<<(B_ * N_) / 64, 128, 0, stream>>>(qb, dotsT, woutb, b_out, out);
}
